// mTAND_43911745635032
// MI455X (gfx1250) — compile-verified, matrix operand order checked by reading, not run on hardware
//
#include <hip/hip_runtime.h>
#include <hip/hip_bf16.h>
#include <math.h>

typedef __attribute__((ext_vector_type(16))) _Float16 v16h;
typedef __attribute__((ext_vector_type(8)))  float    v8f;
typedef _Float16 h16_t;

// ---------------------------------------------------------------------------
// WMMA helpers (CDNA5 v_wmma_f32_16x16x32_f16, wave32)
// A-frag: lane = row (lane&15), K striping per ISA 7.12.2
// B-frag: weights stored row-major (out,in) == B^T, lane = output col, same K striping
// ---------------------------------------------------------------------------
__device__ __forceinline__ v16h frag_ab(const h16_t* base, int ld) {
  int lane = threadIdx.x & 31;
  const h16_t* p = base + (lane & 15) * ld;
  int half8 = (lane & 16) ? 8 : 0;
  v16h f;
#pragma unroll
  for (int v = 0; v < 8; ++v) {
    int k0 = ((v >> 2) << 4) + half8 + ((v & 3) << 1);
    f[2 * v]     = p[k0];
    f[2 * v + 1] = p[k0 + 1];
  }
  return f;
}

__device__ __forceinline__ v8f wmma_mac(v16h a, v16h b, v8f c) {
  return __builtin_amdgcn_wmma_f32_16x16x32_f16(false, a, false, b, (short)0, c,
                                                false, false);
}

// ---------------------------------------------------------------------------
// 0: all f32 -> f16 weight conversions fused into one launch
//    regions: qw(16384) kw(16384) ow(8192) wih(49152) whh(49152) = 139264
// ---------------------------------------------------------------------------
__global__ void k_cvt_all(const float* __restrict__ qw, const float* __restrict__ kw,
                          const float* __restrict__ ow, const float* __restrict__ wih,
                          const float* __restrict__ whh,
                          h16_t* __restrict__ qw16, h16_t* __restrict__ kw16,
                          h16_t* __restrict__ ow16, h16_t* __restrict__ wih16,
                          h16_t* __restrict__ whh16) {
  int i = blockIdx.x * blockDim.x + threadIdx.x;  // 544*256 == 139264 exactly
  if (i < 16384) { qw16[i] = (h16_t)qw[i]; return; }
  i -= 16384;
  if (i < 16384) { kw16[i] = (h16_t)kw[i]; return; }
  i -= 16384;
  if (i < 8192)  { ow16[i] = (h16_t)ow[i]; return; }
  i -= 8192;
  if (i < 49152) { wih16[i] = (h16_t)wih[i]; return; }
  i -= 49152;
  whh16[i] = (h16_t)whh[i];
}

// ---------------------------------------------------------------------------
// 1: time embedding for 4096 key rows + 128 query rows  (f16 output)
// ---------------------------------------------------------------------------
__global__ void k_embed(const float* __restrict__ ts, const float* __restrict__ pw,
                        const float* __restrict__ pb, const float* __restrict__ lw,
                        const float* __restrict__ lb, h16_t* __restrict__ keye,
                        h16_t* __restrict__ qe) {
  int idx = blockIdx.x * blockDim.x + threadIdx.x;  // 4224*128 exactly
  int r = idx >> 7, c = idx & 127;
  float t = (r < 4096) ? ts[r] : (float)(r - 4096) * (1.0f / 127.0f);
  float v = (c == 0) ? (t * lw[0] + lb[0]) : sinf(t * pw[c - 1] + pb[c - 1]);
  if (r < 4096) keye[idx] = (h16_t)v;
  else          qe[(r - 4096) * 128 + c] = (h16_t)v;
}

// ---------------------------------------------------------------------------
// 2: E->E projection (WMMA) + bias + LayerNorm, for keys (rows<4096) / queries
// ---------------------------------------------------------------------------
__global__ void k_proj_ln(const h16_t* __restrict__ keye, const h16_t* __restrict__ qe,
                          const h16_t* __restrict__ kw16, const h16_t* __restrict__ qw16,
                          const float* __restrict__ kb, const float* __restrict__ qb,
                          const float* __restrict__ kg, const float* __restrict__ kbe,
                          const float* __restrict__ qg, const float* __restrict__ qbe,
                          h16_t* __restrict__ k16, h16_t* __restrict__ q16) {
  __shared__ float tile[16 * 128];
  __shared__ float mu[16], rs[16];
  int r0 = blockIdx.x * 16;
  bool isQ = (r0 >= 4096);
  const h16_t* Abase = isQ ? (qe + (r0 - 4096) * 128) : (keye + r0 * 128);
  const h16_t* W     = isQ ? qw16 : kw16;
  const float* bias  = isQ ? qb : kb;
  int wave = threadIdx.x >> 5;
  int n0 = wave * 16;
  v8f acc = {};
#pragma unroll
  for (int kc = 0; kc < 128; kc += 32)
    acc = wmma_mac(frag_ab(Abase + kc, 128), frag_ab(W + n0 * 128 + kc, 128), acc);
  {
    int lane = threadIdx.x & 31;
    int n = n0 + (lane & 15);
    int mb = (lane & 16) ? 8 : 0;
    float bv = bias[n];
#pragma unroll
    for (int rr = 0; rr < 8; ++rr) tile[(mb + rr) * 128 + n] = acc[rr] + bv;
  }
  __syncthreads();
  if (threadIdx.x < 16) {
    float s = 0.f;
    for (int c = 0; c < 128; ++c) s += tile[threadIdx.x * 128 + c];
    float m = s * (1.0f / 128.0f);
    float v = 0.f;
    for (int c = 0; c < 128; ++c) { float d = tile[threadIdx.x * 128 + c] - m; v += d * d; }
    mu[threadIdx.x] = m;
    rs[threadIdx.x] = rsqrtf(v * (1.0f / 128.0f) + 1e-5f);
  }
  __syncthreads();
  const float* g  = isQ ? qg : kg;
  const float* be = isQ ? qbe : kbe;
  h16_t* outp = isQ ? (q16 + (r0 - 4096) * 128) : (k16 + r0 * 128);
#pragma unroll
  for (int i = 0; i < 8; ++i) {
    int e = threadIdx.x * 8 + i;
    int rr = e >> 7, c = e & 127;
    outp[rr * 128 + c] = (h16_t)((tile[e] - mu[rr]) * rs[rr] * g[c] + be[c]);
  }
}

// ---------------------------------------------------------------------------
// 3: scores[b][q][t] = q . k / sqrt(128)  (WMMA, 2048 wave-tiles)
// ---------------------------------------------------------------------------
__global__ void k_scores(const h16_t* __restrict__ q16, const h16_t* __restrict__ k16,
                         float* __restrict__ scores) {
  int gt = blockIdx.x * 8 + (threadIdx.x >> 5);  // 0..2047
  int b = gt >> 8;
  int rem = gt & 255;
  int mt = rem >> 5, nt = rem & 31;
  const h16_t* Ab = q16 + mt * 16 * 128;
  const h16_t* Bb = k16 + (b * 512 + nt * 16) * 128;
  v8f acc = {};
#pragma unroll
  for (int kc = 0; kc < 128; kc += 32)
    acc = wmma_mac(frag_ab(Ab + kc, 128), frag_ab(Bb + kc, 128), acc);
  int lane = threadIdx.x & 31;
  float* C = scores + (b * 128 + mt * 16) * 512 + nt * 16 + (lane & 15);
  int mb = (lane & 16) ? 8 : 0;
  const float s = 0.08838834764831845f;  // 1/sqrt(128)
#pragma unroll
  for (int rr = 0; rr < 8; ++rr) C[(mb + rr) * 512] = acc[rr] * s;
}

// ---------------------------------------------------------------------------
// 4: per-channel masked softmax over keys + weighted sums -> att (B,NT,64)
//    one wave per (b, q, c); 16 keys per lane, shfl_xor reductions
// ---------------------------------------------------------------------------
__global__ void k_attn(const float* __restrict__ scores, const float* __restrict__ input,
                       const float* __restrict__ mask, float* __restrict__ att) {
  int gid = blockIdx.x * 8 + (threadIdx.x >> 5);  // 0..32767
  int lane = threadIdx.x & 31;
  int b = gid >> 12;
  int rem = gid & 4095;
  int qi = rem >> 5, c = rem & 31;
  const float* sp = scores + (b * 128 + qi) * 512;
  const float* mp = mask + (b * 32 + c) * 512;
  const float* xp = input + (b * 32 + c) * 512;
  float lg[16], xv[16], mv[16];
  float mx = -3.0e38f;
#pragma unroll
  for (int i = 0; i < 16; ++i) {
    int t = lane + 32 * i;
    float m = 1.0f - mp[t];              // m in {0,1}
    mv[i] = m;
    xv[i] = xp[t] * m;                   // x_in * (m>0)
    float l = (m == 0.0f) ? -1e9f : sp[t];
    lg[i] = l;
    mx = fmaxf(mx, l);
  }
#pragma unroll
  for (int o = 16; o > 0; o >>= 1) mx = fmaxf(mx, __shfl_xor(mx, o, 32));
  float se = 0.f, sx = 0.f, sm = 0.f;
#pragma unroll
  for (int i = 0; i < 16; ++i) {
    float e = expf(lg[i] - mx);
    se += e; sx += e * xv[i]; sm += e * mv[i];
  }
#pragma unroll
  for (int o = 16; o > 0; o >>= 1) {
    se += __shfl_xor(se, o, 32);
    sx += __shfl_xor(sx, o, 32);
    sm += __shfl_xor(sm, o, 32);
  }
  if (lane == 0) {
    float inv = 1.0f / se;
    float* ap = att + (b * 128 + qi) * 64;
    ap[c]      = sx * inv;
    ap[c + 32] = sm * inv;
  }
}

// ---------------------------------------------------------------------------
// 5: LN(att) -> out_w GEMM (K=64, WMMA) -> LN -> out16 (f16)
// ---------------------------------------------------------------------------
__global__ void k_outproj(const float* __restrict__ att, const float* __restrict__ ag,
                          const float* __restrict__ ab, const h16_t* __restrict__ ow16,
                          const float* __restrict__ ob, const float* __restrict__ og,
                          const float* __restrict__ obe, h16_t* __restrict__ out16) {
  __shared__ float t1[16 * 64];
  __shared__ h16_t x2[16 * 64];
  __shared__ float t2[16 * 128];
  __shared__ float mu[16], rs[16];
  int r0 = blockIdx.x * 16;
#pragma unroll
  for (int i = 0; i < 4; ++i) {
    int e = threadIdx.x * 4 + i;
    t1[e] = att[r0 * 64 + e];
  }
  __syncthreads();
  if (threadIdx.x < 16) {
    float s = 0.f;
    for (int c = 0; c < 64; ++c) s += t1[threadIdx.x * 64 + c];
    float m = s * (1.0f / 64.0f);
    float v = 0.f;
    for (int c = 0; c < 64; ++c) { float d = t1[threadIdx.x * 64 + c] - m; v += d * d; }
    mu[threadIdx.x] = m;
    rs[threadIdx.x] = rsqrtf(v * (1.0f / 64.0f) + 1e-5f);
  }
  __syncthreads();
#pragma unroll
  for (int i = 0; i < 4; ++i) {
    int e = threadIdx.x * 4 + i;
    int rr = e >> 6, c = e & 63;
    x2[e] = (h16_t)((t1[e] - mu[rr]) * rs[rr] * ag[c] + ab[c]);
  }
  __syncthreads();
  int wave = threadIdx.x >> 5;
  int n0 = wave * 16;
  v8f acc = {};
#pragma unroll
  for (int kc = 0; kc < 64; kc += 32)
    acc = wmma_mac(frag_ab(x2 + kc, 64), frag_ab(ow16 + n0 * 64 + kc, 64), acc);
  {
    int lane = threadIdx.x & 31;
    int n = n0 + (lane & 15);
    int mb = (lane & 16) ? 8 : 0;
    float bv = ob[n];
#pragma unroll
    for (int rr = 0; rr < 8; ++rr) t2[(mb + rr) * 128 + n] = acc[rr] + bv;
  }
  __syncthreads();
  if (threadIdx.x < 16) {
    float s = 0.f;
    for (int c = 0; c < 128; ++c) s += t2[threadIdx.x * 128 + c];
    float m = s * (1.0f / 128.0f);
    float v = 0.f;
    for (int c = 0; c < 128; ++c) { float d = t2[threadIdx.x * 128 + c] - m; v += d * d; }
    mu[threadIdx.x] = m;
    rs[threadIdx.x] = rsqrtf(v * (1.0f / 128.0f) + 1e-5f);
  }
  __syncthreads();
#pragma unroll
  for (int i = 0; i < 8; ++i) {
    int e = threadIdx.x * 8 + i;
    int rr = e >> 7, c = e & 127;
    out16[(r0 + rr) * 128 + c] = (h16_t)((t2[e] - mu[rr]) * rs[rr] * og[c] + obe[c]);
  }
}

// ---------------------------------------------------------------------------
// 6: gi = out @ W_ih^T + b_ih, remapped to time-major [t][b][384]
// ---------------------------------------------------------------------------
__global__ void k_gi(const h16_t* __restrict__ out16, const h16_t* __restrict__ wih16,
                     const float* __restrict__ bih, float* __restrict__ gi) {
  int gt = blockIdx.x * 8 + (threadIdx.x >> 5);  // 0..1535
  int mt = gt / 24, ct = gt % 24;
  const h16_t* Ab = out16 + mt * 16 * 128;
  const h16_t* Bb = wih16 + ct * 16 * 128;
  v8f acc = {};
#pragma unroll
  for (int kc = 0; kc < 128; kc += 32)
    acc = wmma_mac(frag_ab(Ab + kc, 128), frag_ab(Bb + kc, 128), acc);
  int lane = threadIdx.x & 31;
  int n = ct * 16 + (lane & 15);
  int mb = (lane & 16) ? 8 : 0;
  float bv = bih[n];
#pragma unroll
  for (int rr = 0; rr < 8; ++rr) {
    int r = mt * 16 + mb + rr;       // row in (b, nt) order
    int bb = r >> 7, nt = r & 127;
    gi[(nt * 8 + bb) * 384 + n] = acc[rr] + bv;
  }
}

// ---------------------------------------------------------------------------
// 7: GRU scan — single persistent workgroup (32 waves).
//    * waves 0..23 hold one 16-col tile of W_hh as 4 register-resident B-frags
//    * h kept in LDS (f32 for update, zero-padded f16 for WMMA)
//    * gi for step t+1 is double-buffered into LDS with
//      GLOBAL_LOAD_ASYNC_TO_LDS_B128 (ASYNCcnt) to hide L2 latency behind
//      the serial recurrence; retired with s_wait_asynccnt before the barrier
// ---------------------------------------------------------------------------
__global__ void __launch_bounds__(1024) k_gru(const float* __restrict__ gi,
                                              const h16_t* __restrict__ whh16,
                                              const float* __restrict__ bhh,
                                              float* __restrict__ hlast) {
  __shared__ float h32[1024];        // h, f32, (b*128 + j)
  __shared__ h16_t hh[16 * 128];     // h, f16, padded to 16 rows for WMMA
  __shared__ float gacc[8 * 384];    // gh + bhh
  __shared__ float gibuf[2][8 * 384];// double-buffered gi slab (12KB each)
  int tid = threadIdx.x;
  int wave = tid >> 5, lane = tid & 31;
  h32[tid] = 0.f;
  hh[tid] = (h16_t)0.f;
  hh[tid + 1024] = (h16_t)0.f;       // rows 8..15 stay zero forever
  // prefill gi(t=0) into gibuf[0]: 3072 floats = 768 x b128
  if (tid < 768) {
    unsigned laddr = (unsigned)(size_t)(&gibuf[0][tid * 4]);
    const float* gsrc = gi + tid * 4;
    asm volatile("global_load_async_to_lds_b128 %0, %1, off"
                 :: "v"(laddr), "v"(gsrc) : "memory");
  }
  asm volatile("s_wait_asynccnt 0x0" ::: "memory");
  __syncthreads();
  v16h bf0 = {}, bf1 = {}, bf2 = {}, bf3 = {};
  if (wave < 24) {
    const h16_t* Bb = whh16 + wave * 16 * 128;
    bf0 = frag_ab(Bb + 0, 128);
    bf1 = frag_ab(Bb + 32, 128);
    bf2 = frag_ab(Bb + 64, 128);
    bf3 = frag_ab(Bb + 96, 128);
  }
  int b = tid >> 7, j = tid & 127;
  for (int t = 0; t < 128; ++t) {
    // kick off async prefetch of gi(t+1) into the other buffer
    if (t + 1 < 128 && tid < 768) {
      unsigned laddr = (unsigned)(size_t)(&gibuf[(t + 1) & 1][tid * 4]);
      const float* gsrc = gi + (t + 1) * 3072 + tid * 4;
      asm volatile("global_load_async_to_lds_b128 %0, %1, off"
                   :: "v"(laddr), "v"(gsrc) : "memory");
    }
    if (wave < 24) {
      v8f acc = {};
      acc = wmma_mac(frag_ab(hh + 0, 128), bf0, acc);
      acc = wmma_mac(frag_ab(hh + 32, 128), bf1, acc);
      acc = wmma_mac(frag_ab(hh + 64, 128), bf2, acc);
      acc = wmma_mac(frag_ab(hh + 96, 128), bf3, acc);
      if (!(lane & 16)) {                    // rows 0..7 are the 8 batches
        int n = wave * 16 + (lane & 15);
        float bv = bhh[n];
#pragma unroll
        for (int rr = 0; rr < 8; ++rr) gacc[rr * 384 + n] = acc[rr] + bv;
      }
    }
    __syncthreads();
    const float* gp = &gibuf[t & 1][b * 384];
    float r = 1.0f / (1.0f + expf(-(gp[j] + gacc[b * 384 + j])));
    float z = 1.0f / (1.0f + expf(-(gp[j + 128] + gacc[b * 384 + j + 128])));
    float n = tanhf(gp[j + 256] + r * gacc[b * 384 + j + 256]);
    float hn = (1.0f - z) * n + z * h32[tid];
    h32[tid] = hn;                // only this thread reads h32[tid]
    hh[tid] = (h16_t)hn;          // consumed only after the barrier below
    asm volatile("s_wait_asynccnt 0x0" ::: "memory");
    __syncthreads();
  }
  hlast[tid] = h32[tid];
}

// ---------------------------------------------------------------------------
// 8: classifier MLP (f32 VALU; 16 final outputs)
// ---------------------------------------------------------------------------
__global__ void k_cls(const float* __restrict__ hlast,
                      const float* __restrict__ c1w, const float* __restrict__ c1b,
                      const float* __restrict__ c2w, const float* __restrict__ c2b,
                      const float* __restrict__ c3w, const float* __restrict__ c3b,
                      float* __restrict__ outp) {
  __shared__ float hb[1024];
  __shared__ float b1[1024];
  __shared__ float b2[1024];
  int tid = threadIdx.x;  // 256
#pragma unroll
  for (int i = 0; i < 4; ++i) hb[tid * 4 + i] = hlast[tid * 4 + i];
  __syncthreads();
#pragma unroll
  for (int i = 0; i < 4; ++i) {
    int o = tid + i * 256;
    int b = o >> 7, j = o & 127;
    float s = c1b[j];
    for (int k = 0; k < 128; ++k) s += hb[b * 128 + k] * c1w[j * 128 + k];
    b1[b * 128 + j] = fmaxf(s, 0.f);
  }
  __syncthreads();
#pragma unroll
  for (int i = 0; i < 4; ++i) {
    int o = tid + i * 256;
    int b = o >> 7, j = o & 127;
    float s = c2b[j];
    for (int k = 0; k < 128; ++k) s += b1[b * 128 + k] * c2w[j * 128 + k];
    b2[b * 128 + j] = fmaxf(s, 0.f);
  }
  __syncthreads();
  if (tid < 16) {
    int b = tid >> 1, s2 = tid & 1;
    float s = c3b[s2];
    for (int k = 0; k < 128; ++k) s += b2[b * 128 + k] * c3w[s2 * 128 + k];
    outp[b * 2 + s2] = s;
  }
}

// ---------------------------------------------------------------------------
extern "C" void kernel_launch(void* const* d_in, const int* in_sizes, int n_in,
                              void* d_out, int out_size, void* d_ws, size_t ws_size,
                              hipStream_t stream) {
  (void)in_sizes; (void)n_in; (void)out_size; (void)ws_size;
  const float* input = (const float*)d_in[0];
  const float* mask  = (const float*)d_in[1];
  const float* ts    = (const float*)d_in[2];
  const float* pw    = (const float*)d_in[3];
  const float* pb    = (const float*)d_in[4];
  const float* lw    = (const float*)d_in[5];
  const float* lb    = (const float*)d_in[6];
  const float* qw    = (const float*)d_in[7];
  const float* qb    = (const float*)d_in[8];
  const float* qg    = (const float*)d_in[9];
  const float* qbe   = (const float*)d_in[10];
  const float* kw    = (const float*)d_in[11];
  const float* kb    = (const float*)d_in[12];
  const float* kg    = (const float*)d_in[13];
  const float* kbe   = (const float*)d_in[14];
  const float* ag    = (const float*)d_in[15];
  const float* ab    = (const float*)d_in[16];
  const float* ow    = (const float*)d_in[17];
  const float* ob    = (const float*)d_in[18];
  const float* og    = (const float*)d_in[19];
  const float* obe   = (const float*)d_in[20];
  const float* wih   = (const float*)d_in[21];
  const float* whh   = (const float*)d_in[22];
  const float* bih   = (const float*)d_in[23];
  const float* bhh   = (const float*)d_in[24];
  const float* c1w   = (const float*)d_in[25];
  const float* c1b   = (const float*)d_in[26];
  const float* c2w   = (const float*)d_in[27];
  const float* c2b   = (const float*)d_in[28];
  const float* c3w   = (const float*)d_in[29];
  const float* c3b   = (const float*)d_in[30];

  char* ws = (char*)d_ws;
  size_t off = 0;
  auto alloc = [&](size_t bytes) {
    char* p = ws + off;
    off += (bytes + 255) & ~(size_t)255;
    return p;
  };
  h16_t* qw16  = (h16_t*)alloc(16384 * 2);
  h16_t* kw16  = (h16_t*)alloc(16384 * 2);
  h16_t* ow16  = (h16_t*)alloc(8192 * 2);
  h16_t* wih16 = (h16_t*)alloc(49152 * 2);
  h16_t* whh16 = (h16_t*)alloc(49152 * 2);
  h16_t* keye  = (h16_t*)alloc(4096 * 128 * 2);
  h16_t* qe    = (h16_t*)alloc(128 * 128 * 2);
  h16_t* k16   = (h16_t*)alloc(4096 * 128 * 2);
  h16_t* q16   = (h16_t*)alloc(128 * 128 * 2);
  float* scores = (float*)alloc((size_t)8 * 128 * 512 * 4);
  float* att    = (float*)alloc((size_t)8 * 128 * 64 * 4);
  h16_t* out16  = (h16_t*)alloc((size_t)1024 * 128 * 2);
  float* gi     = (float*)alloc((size_t)128 * 8 * 384 * 4);
  float* hlast  = (float*)alloc(1024 * 4);

  k_cvt_all<<<544, 256, 0, stream>>>(qw, kw, ow, wih, whh,
                                     qw16, kw16, ow16, wih16, whh16);
  k_embed<<<2112, 256, 0, stream>>>(ts, pw, pb, lw, lb, keye, qe);
  k_proj_ln<<<264, 256, 0, stream>>>(keye, qe, kw16, qw16, kb, qb, kg, kbe, qg, qbe,
                                     k16, q16);
  k_scores<<<256, 256, 0, stream>>>(q16, k16, scores);
  k_attn<<<4096, 256, 0, stream>>>(scores, input, mask, att);
  k_outproj<<<64, 256, 0, stream>>>(att, ag, ab, ow16, ob, og, obe, out16);
  k_gi<<<192, 256, 0, stream>>>(out16, wih16, bih, gi);
  k_gru<<<1, 1024, 0, stream>>>(gi, whh16, bhh, hlast);
  k_cls<<<1, 256, 0, stream>>>(hlast, c1w, c1b, c2w, c2b, c3w, c3b, (float*)d_out);
}
